// MidmLMHeadModelWrapper_27135603376467
// MI455X (gfx1250) — matrix operand order repaired and checked by reading, not compiled, not run on hardware
//
#include <hip/hip_runtime.h>
#include <hip/hip_bf16.h>

// Problem constants (match reference)
#define BB     2
#define QQ     512
#define DD     4096
#define HH     32
#define HDIM   128
#define MAXLEN 4096
#define ROT    64

typedef __bf16 bf16_t;
typedef bf16_t v16bf __attribute__((ext_vector_type(16)));
typedef float  v8f   __attribute__((ext_vector_type(8)));

union Frag {
    v16bf v;
    unsigned short us[16];
};

__device__ __forceinline__ unsigned short f32_to_bf16_rne(float f) {
    unsigned int u = __float_as_uint(f);
    u += 0x7FFFu + ((u >> 16) & 1u);   // round to nearest even
    return (unsigned short)(u >> 16);
}

__device__ __forceinline__ v8f wmma_bf16(const Frag& a, const Frag& b, v8f c) {
    return __builtin_amdgcn_wmma_f32_16x16x32_bf16(
        /*neg_a=*/false, a.v, /*neg_b=*/false, b.v,
        /*c_mod=*/(short)0, c, /*reuse_a=*/false, /*reuse_b=*/false);
}

// ---------------------------------------------------------------------------
// Generic f32-in / f32-out GEMM with bf16 WMMA compute.  C = A(MxK) * B(KxN) + bias
// Block tile 128x128, 8 waves (4Mx2N), each wave 32x64 (2x4 WMMA accumulators).
// ---------------------------------------------------------------------------
__global__ __launch_bounds__(256) void gemm_bf16_kernel(
    const float* __restrict__ A, const float* __restrict__ Bm,
    const float* __restrict__ bias, float* __restrict__ C,
    int M, int N, int K)
{
    __shared__ alignas(16) unsigned short lA[128 * 32];   // row-major, stride 32
    __shared__ alignas(16) unsigned short lB[128 * 32];   // B^T: n-major, stride 32

    const int bn   = blockIdx.x * 128;
    const int bm   = blockIdx.y * 128;
    const int tid  = threadIdx.x;
    const int lane = tid & 31;
    const int wid  = tid >> 5;
    const int wm   = (wid >> 1) * 32;   // wave M offset in block: 0/32/64/96
    const int wn   = (wid & 1) * 64;    // wave N offset in block: 0/64

    v8f acc[2][4];
    const v8f vzero = {0.f,0.f,0.f,0.f,0.f,0.f,0.f,0.f};
#pragma unroll
    for (int mt = 0; mt < 2; ++mt)
#pragma unroll
        for (int nt = 0; nt < 4; ++nt) acc[mt][nt] = vzero;

    const int arow = tid >> 1;            // 0..127
    const int acol = (tid & 1) * 16;      // 0/16
    const int bkr  = tid >> 3;            // 0..31
    const int bnc  = (tid & 7) * 16;      // 0..112

    for (int kc = 0; kc < K; kc += 32) {
        // ---- stage A tile (128x32) as bf16, row-major ----
        {
            const float* src = A + (size_t)(bm + arow) * K + kc + acol;
            unsigned short* dst = lA + arow * 32 + acol;
#pragma unroll
            for (int i = 0; i < 16; i += 4) {
                float4 f = *(const float4*)(src + i);
                dst[i + 0] = f32_to_bf16_rne(f.x);
                dst[i + 1] = f32_to_bf16_rne(f.y);
                dst[i + 2] = f32_to_bf16_rne(f.z);
                dst[i + 3] = f32_to_bf16_rne(f.w);
            }
        }
        // ---- stage B^T tile (128n x 32k) as bf16 ----
        {
            const float* src = Bm + (size_t)(kc + bkr) * N + bn + bnc;
#pragma unroll
            for (int i = 0; i < 16; i += 4) {
                float4 f = *(const float4*)(src + i);
                lB[(bnc + i + 0) * 32 + bkr] = f32_to_bf16_rne(f.x);
                lB[(bnc + i + 1) * 32 + bkr] = f32_to_bf16_rne(f.y);
                lB[(bnc + i + 2) * 32 + bkr] = f32_to_bf16_rne(f.z);
                lB[(bnc + i + 3) * 32 + bkr] = f32_to_bf16_rne(f.w);
            }
        }
        __syncthreads();

        // ---- WMMA fragments from LDS ----
        Frag af[2];
        {
            const int base8 = (lane < 16) ? 0 : 8;
#pragma unroll
            for (int mt = 0; mt < 2; ++mt) {
                const unsigned short* p = lA + (wm + mt * 16 + (lane & 15)) * 32 + base8;
                ((uint4*)af[mt].us)[0] = *(const uint4*)p;        // K = base..base+7
                ((uint4*)af[mt].us)[1] = *(const uint4*)(p + 16); // K = base+16..base+23
            }
        }
        const int bbase8 = (lane < 16) ? 0 : 8;
#pragma unroll
        for (int nt = 0; nt < 4; ++nt) {
            Frag bfg;
            const unsigned short* p = lB + (wn + nt * 16 + (lane & 15)) * 32 + bbase8;
            ((uint4*)bfg.us)[0] = *(const uint4*)p;
            ((uint4*)bfg.us)[1] = *(const uint4*)(p + 16);
#pragma unroll
            for (int mt = 0; mt < 2; ++mt)
                acc[mt][nt] = wmma_bf16(af[mt], bfg, acc[mt][nt]);
        }
        __syncthreads();
    }

    // ---- epilogue: bias + store (C layout: row = r + 8*(lane/16), col = lane%16) ----
    const int nlane = lane & 15;
    const int rbase = (lane < 16) ? 0 : 8;
#pragma unroll
    for (int nt = 0; nt < 4; ++nt) {
        const int gn = bn + wn + nt * 16 + nlane;
        const float bv = bias ? bias[gn] : 0.0f;
#pragma unroll
        for (int mt = 0; mt < 2; ++mt) {
#pragma unroll
            for (int r = 0; r < 8; ++r) {
                const int gm = bm + wm + mt * 16 + rbase + r;
                C[(size_t)gm * N + gn] = acc[mt][nt][r] + bv;
            }
        }
    }
}

// ---------------------------------------------------------------------------
// RoPE + head split.  qkv: (B*Q, 3D) f32.  Outputs q_r/k_new/v_new in
// (B,H,Q,HD) layout.  One block per (b*Q+q, h), 128 threads (one per hd dim).
// ---------------------------------------------------------------------------
__global__ __launch_bounds__(128) void rope_scatter_kernel(
    const float* __restrict__ qkv, const int* __restrict__ pos_ids,
    const float* __restrict__ freqs,
    float* __restrict__ q_r, float* __restrict__ k_new, float* __restrict__ v_new)
{
    const int bq = blockIdx.x;          // 0 .. B*Q-1
    const int h  = blockIdx.y;          // 0 .. H-1
    const int j  = threadIdx.x;         // 0 .. 127
    const int b  = bq / QQ;
    const int q  = bq % QQ;

    const float* row = qkv + (size_t)bq * (3 * DD) + h * HDIM;
    const int pos = pos_ids[bq];
    float c = 0.f, s = 0.f;
    if (j < ROT) {
        const float f = freqs[pos * ROT + j];
        c = __cosf(f);
        s = __sinf(f);
    }
    const size_t oidx = (((size_t)(b * HH + h)) * QQ + q) * HDIM + j;

    // q
    {
        const float t = row[j];
        float val = t;
        if (j < ROT) {
            const float rot = (j < ROT / 2) ? -row[j + ROT / 2] : row[j - ROT / 2];
            val = t * c + rot * s;
        }
        q_r[oidx] = val;
    }
    // k
    {
        const float* rk = row + DD;
        const float t = rk[j];
        float val = t;
        if (j < ROT) {
            const float rot = (j < ROT / 2) ? -rk[j + ROT / 2] : rk[j - ROT / 2];
            val = t * c + rot * s;
        }
        k_new[oidx] = val;
    }
    // v (no rope)
    v_new[oidx] = row[2 * DD + j];
}

// ---------------------------------------------------------------------------
// Flash-style attention.  Block = (qblock of 128 rows, h, b); 8 waves, each
// owns 16 q rows.  Per 32-key block: cooperative K / V^T staging into LDS,
// bf16 WMMA for q.k^T and P.V, online softmax in f32.
// ---------------------------------------------------------------------------
__global__ __launch_bounds__(256) void attn_kernel(
    const float* __restrict__ q_r, const float* __restrict__ k_new,
    const float* __restrict__ v_new,
    const float* __restrict__ past_key, const float* __restrict__ past_value,
    const float* __restrict__ amask,
    const int* __restrict__ cur_step_p, const int* __restrict__ layer_idx_p,
    float* __restrict__ attn_out)
{
    __shared__ alignas(16) unsigned short lK[32 * 128];       // key-major, stride 128
    __shared__ alignas(16) unsigned short lVT[128 * 32];      // hd-major,  stride 32
    __shared__ alignas(16) unsigned short lP[8][16 * 32];     // per-wave P tile

    const int qblk = blockIdx.x;
    const int h    = blockIdx.y;
    const int b    = blockIdx.z;
    const int tid  = threadIdx.x;
    const int lane = tid & 31;
    const int wid  = tid >> 5;

    const int   cur     = cur_step_p[0];
    const float lscale  = (float)(layer_idx_p[0] + 1);
    const float invsqrt = 0.08838834764831845f;   // 1/sqrt(128)

    const int q0 = qblk * 128 + wid * 16;

    // ---- resident Q fragments (16 x 128 bf16, 4 K-chunks) ----
    Frag qf[4];
    {
        const int row = q0 + (lane & 15);
        const float* qptr = q_r + ((size_t)(b * HH + h) * QQ + row) * HDIM;
        const int base8 = (lane < 16) ? 0 : 8;
#pragma unroll
        for (int c = 0; c < 4; ++c) {
            const float* p = qptr + c * 32 + base8;
            float4 f0 = *(const float4*)(p + 0);
            float4 f1 = *(const float4*)(p + 4);
            float4 f2 = *(const float4*)(p + 16);
            float4 f3 = *(const float4*)(p + 20);
            qf[c].us[0]  = f32_to_bf16_rne(f0.x); qf[c].us[1]  = f32_to_bf16_rne(f0.y);
            qf[c].us[2]  = f32_to_bf16_rne(f0.z); qf[c].us[3]  = f32_to_bf16_rne(f0.w);
            qf[c].us[4]  = f32_to_bf16_rne(f1.x); qf[c].us[5]  = f32_to_bf16_rne(f1.y);
            qf[c].us[6]  = f32_to_bf16_rne(f1.z); qf[c].us[7]  = f32_to_bf16_rne(f1.w);
            qf[c].us[8]  = f32_to_bf16_rne(f2.x); qf[c].us[9]  = f32_to_bf16_rne(f2.y);
            qf[c].us[10] = f32_to_bf16_rne(f2.z); qf[c].us[11] = f32_to_bf16_rne(f2.w);
            qf[c].us[12] = f32_to_bf16_rne(f3.x); qf[c].us[13] = f32_to_bf16_rne(f3.y);
            qf[c].us[14] = f32_to_bf16_rne(f3.z); qf[c].us[15] = f32_to_bf16_rne(f3.w);
        }
    }

    v8f O[8];
    const v8f vzero = {0.f,0.f,0.f,0.f,0.f,0.f,0.f,0.f};
#pragma unroll
    for (int t = 0; t < 8; ++t) O[t] = vzero;
    float mrow[8], lrow[8];
#pragma unroll
    for (int r = 0; r < 8; ++r) { mrow[r] = -1.0e30f; lrow[r] = 0.0f; }

    const float* kc_base = past_key   + ((size_t)(b * HH + h)) * MAXLEN * HDIM;
    const float* vc_base = past_value + ((size_t)(b * HH + h)) * MAXLEN * HDIM;
    const float* kn_base = k_new + ((size_t)(b * HH + h)) * QQ * HDIM;
    const float* vn_base = v_new + ((size_t)(b * HH + h)) * QQ * HDIM;

    const int nlane = lane & 15;
    const int rbase = (lane < 16) ? 0 : 8;
    const int base16 = (lane < 16) ? 0 : 16;
    const int base8  = (lane < 16) ? 0 : 8;

    for (int kb = 0; kb < MAXLEN; kb += 32) {
        // ---- cooperative staging of K (key-major) and V^T (hd-major) ----
        {
            const int r  = tid >> 3;           // key row 0..31
            const int c0 = (tid & 7) * 16;     // hd chunk
            const int kk = kb + r;
            const float* ksrc;
            const float* vsrc;
            if (kk >= cur && kk < cur + QQ) {
                ksrc = kn_base + (size_t)(kk - cur) * HDIM;
                vsrc = vn_base + (size_t)(kk - cur) * HDIM;
            } else {
                ksrc = kc_base + (size_t)kk * HDIM;
                vsrc = vc_base + (size_t)kk * HDIM;
            }
#pragma unroll
            for (int i = 0; i < 16; i += 4) {
                float4 f = *(const float4*)(ksrc + c0 + i);
                lK[r * 128 + c0 + i + 0] = f32_to_bf16_rne(f.x);
                lK[r * 128 + c0 + i + 1] = f32_to_bf16_rne(f.y);
                lK[r * 128 + c0 + i + 2] = f32_to_bf16_rne(f.z);
                lK[r * 128 + c0 + i + 3] = f32_to_bf16_rne(f.w);
            }
#pragma unroll
            for (int i = 0; i < 16; i += 4) {
                float4 f = *(const float4*)(vsrc + c0 + i);
                lVT[(c0 + i + 0) * 32 + r] = f32_to_bf16_rne(f.x);
                lVT[(c0 + i + 1) * 32 + r] = f32_to_bf16_rne(f.y);
                lVT[(c0 + i + 2) * 32 + r] = f32_to_bf16_rne(f.z);
                lVT[(c0 + i + 3) * 32 + r] = f32_to_bf16_rne(f.w);
            }
        }
        __syncthreads();

        // ---- S = q . k^T  (two 16x16 key tiles) ----
        v8f sacc[2];
#pragma unroll
        for (int nt = 0; nt < 2; ++nt) {
            sacc[nt] = vzero;
            const unsigned short* krow = lK + (nt * 16 + nlane) * 128 + base8;
#pragma unroll
            for (int c = 0; c < 4; ++c) {
                Frag kf;
                const unsigned short* p = krow + c * 32;
                ((uint4*)kf.us)[0] = *(const uint4*)p;
                ((uint4*)kf.us)[1] = *(const uint4*)(p + 16);
                sacc[nt] = wmma_bf16(qf[c], kf, sacc[nt]);
            }
        }

        // ---- online softmax update (rows = rbase+r for this half-wave) ----
        float pv0[8], pv1[8];
#pragma unroll
        for (int r = 0; r < 8; ++r) {
            const int gq = q0 + rbase + r;
            const size_t mrowofs = ((size_t)b * QQ + gq) * MAXLEN + kb;
            float s0 = sacc[0][r] * invsqrt + amask[mrowofs + nlane] * lscale;
            float s1 = sacc[1][r] * invsqrt + amask[mrowofs + 16 + nlane] * lscale;
            float mx = fmaxf(s0, s1);
#pragma unroll
            for (int off = 1; off < 16; off <<= 1)
                mx = fmaxf(mx, __shfl_xor(mx, off, 32));
            const float mnew = fmaxf(mrow[r], mx);
            const float corr = __expf(mrow[r] - mnew);
            const float p0 = __expf(s0 - mnew);
            const float p1 = __expf(s1 - mnew);
            float rs = p0 + p1;
#pragma unroll
            for (int off = 1; off < 16; off <<= 1)
                rs += __shfl_xor(rs, off, 32);
            lrow[r] = lrow[r] * corr + rs;
            mrow[r] = mnew;
            pv0[r] = p0;
            pv1[r] = p1;
#pragma unroll
            for (int t = 0; t < 8; ++t) O[t][r] *= corr;
        }

        // ---- P -> per-wave LDS (bf16, row-major stride 32) ----
        unsigned short* pw = lP[wid];
#pragma unroll
        for (int r = 0; r < 8; ++r) {
            const int rowm = rbase + r;
            pw[rowm * 32 + nlane]      = f32_to_bf16_rne(pv0[r]);
            pw[rowm * 32 + 16 + nlane] = f32_to_bf16_rne(pv1[r]);
        }
        // same-wave LDS RAW: DS ops are in-order per wave; keep the compiler
        // from reordering and drain the DS counter.
        asm volatile("s_wait_dscnt 0" ::: "memory");
        __builtin_amdgcn_wave_barrier();

        // ---- O += P . V ----
        Frag pf;
        {
            const unsigned short* pp = pw + nlane * 32 + base8;
            ((uint4*)pf.us)[0] = *(const uint4*)pp;
            ((uint4*)pf.us)[1] = *(const uint4*)(pp + 16);
        }
#pragma unroll
        for (int t = 0; t < 8; ++t) {
            Frag vf;
            const unsigned short* vp = lVT + (t * 16 + nlane) * 32 + base8;
            ((uint4*)vf.us)[0] = *(const uint4*)vp;
            ((uint4*)vf.us)[1] = *(const uint4*)(vp + 16);
            O[t] = wmma_bf16(pf, vf, O[t]);
        }
        __syncthreads();
    }

    // ---- epilogue: normalize and store in (B, Q, H, HD) layout ----
#pragma unroll
    for (int r = 0; r < 8; ++r) {
        const float inv = 1.0f / lrow[r];
        const int gq = q0 + rbase + r;
        float* orow = attn_out + ((size_t)(b * QQ + gq) * HH + h) * HDIM;
#pragma unroll
        for (int t = 0; t < 8; ++t)
            orow[t * 16 + nlane] = O[t][r] * inv;
    }
}

// ---------------------------------------------------------------------------
// Launch.  d_in order: hidden, mask, freqs, position_ids, past_key,
// past_value, w_qkv, b_qkv, w_proj, b_proj, current_step, layer_idx
// ---------------------------------------------------------------------------
extern "C" void kernel_launch(void* const* d_in, const int* in_sizes, int n_in,
                              void* d_out, int out_size, void* d_ws, size_t ws_size,
                              hipStream_t stream) {
    const float* hidden     = (const float*)d_in[0];
    const float* amask      = (const float*)d_in[1];
    const float* freqs      = (const float*)d_in[2];
    const int*   pos_ids    = (const int*)  d_in[3];
    const float* past_key   = (const float*)d_in[4];
    const float* past_value = (const float*)d_in[5];
    const float* w_qkv      = (const float*)d_in[6];
    const float* b_qkv      = (const float*)d_in[7];
    const float* w_proj     = (const float*)d_in[8];
    const float* b_proj     = (const float*)d_in[9];
    const int*   cur_step   = (const int*)  d_in[10];
    const int*   layer_idx  = (const int*)  d_in[11];
    float*       out        = (float*)d_out;

    const size_t M   = (size_t)BB * QQ;            // 1024
    const size_t BHQ = (size_t)BB * HH * QQ * HDIM; // 4,194,304 floats

    float* ws       = (float*)d_ws;
    float* qkv      = ws;                      // 1024 x 12288
    float* q_r      = qkv + M * (3 * DD);
    float* k_new    = q_r + BHQ;
    float* v_new    = k_new + BHQ;
    float* attn_o   = v_new + BHQ;

    // 1) QKV projection: (1024 x 4096) @ (4096 x 12288) + bias
    gemm_bf16_kernel<<<dim3(3 * DD / 128, M / 128), 256, 0, stream>>>(
        hidden, w_qkv, b_qkv, qkv, (int)M, 3 * DD, DD);

    // 2) RoPE + head split + v copy
    rope_scatter_kernel<<<dim3(BB * QQ, HH), 128, 0, stream>>>(
        qkv, pos_ids, freqs, q_r, k_new, v_new);

    // 3) Attention over 4096-entry cache (new keys read from k_new/v_new)
    attn_kernel<<<dim3(QQ / 128, HH, BB), 256, 0, stream>>>(
        q_r, k_new, v_new, past_key, past_value, amask,
        cur_step, layer_idx, attn_o);

    // 4) Output projection: (1024 x 4096) @ (4096 x 4096) + bias
    gemm_bf16_kernel<<<dim3(DD / 128, M / 128), 256, 0, stream>>>(
        attn_o, w_proj, b_proj, out, (int)M, DD, DD);
}
